// AttentionBlock_51591147159809
// MI455X (gfx1250) — matrix operand order repaired and checked by reading, not compiled, not run on hardware
//
#include <hip/hip_runtime.h>
#include <hip/hip_bf16.h>

typedef _Float16 half_t;
typedef __attribute__((ext_vector_type(8)))  _Float16 v8h;
typedef __attribute__((ext_vector_type(16))) _Float16 v16h;
typedef __attribute__((ext_vector_type(8)))  float    v8f;

#define HW 4096   // 64*64 spatial
#define CC 512    // channels
#define NB 4      // batch

// ---------------------------------------------------------------------------
// WMMA fragment helpers (layouts per CDNA5 ISA 7.12.2, wave32)
// ---------------------------------------------------------------------------

// A fragment: row-major A[m][k], lda in elements.
// lane&15 = m offset; lane>>4 selects K {0..7,16..23} vs {8..15,24..31}.
__device__ inline v16h load_frag_a(const half_t* __restrict__ A, int lda, int m0, int k0) {
  int lane = threadIdx.x & 31;
  int m = m0 + (lane & 15);
  int koff = (lane >> 4) * 8;
  const half_t* r = A + (size_t)m * lda + k0 + koff;
  union { v16h v; v8h h[2]; } u;
  u.h[0] = *(const v8h*)(r);        // K = koff + 0..7
  u.h[1] = *(const v8h*)(r + 16);   // K = koff + 16..23
  return u.v;
}

// B fragment from K-contiguous storage Bt[n][k] (ldb in elements).
// lane&15 = n; lane>>4 selects K {0..7,16..23} vs {8..15,24..31} (matches A).
__device__ inline v16h load_frag_bt(const half_t* __restrict__ Bt, int ldb, int n0, int k0) {
  int lane = threadIdx.x & 31;
  int n = n0 + (lane & 15);
  int k = k0 + (lane >> 4) * 8;
  const half_t* r = Bt + (size_t)n * ldb + k;
  union { v16h v; v8h h[2]; } u;
  u.h[0] = *(const v8h*)(r);
  u.h[1] = *(const v8h*)(r + 16);
  return u.v;
}

#define WMMA_F16(af, bf, c) \
  __builtin_amdgcn_wmma_f32_16x16x32_f16(false, (af), false, (bf), (short)0, (c), false, false)

// One wave computes a 64(M) x 32(N) tile: 8 accumulators, each A fragment
// feeds two WMMAs (1.5 b128 loads per WMMA instead of 2.5).
__device__ inline void gemm_wave_64x32(const half_t* __restrict__ A, int lda,
                                       const half_t* __restrict__ Bt, int ldb,
                                       int m_blk, int n0, int K, v8f acc[8]) {
  int lane = threadIdx.x & 31;
  for (int k = 0; k < K; k += 32) {
    if (k + 64 < K) {  // speculative L2 prefetch, 2 k-steps ahead
      __builtin_prefetch(Bt + (size_t)(n0 + (lane & 15)) * ldb + k + 64, 0, 3);
      __builtin_prefetch(A + (size_t)(m_blk + lane) * lda + k + 64, 0, 3);
    }
    v16h bf0 = load_frag_bt(Bt, ldb, n0, k);
    v16h bf1 = load_frag_bt(Bt, ldb, n0 + 16, k);
#pragma unroll
    for (int t = 0; t < 4; ++t) {
      v16h af = load_frag_a(A, lda, m_blk + 16 * t, k);
      acc[t]     = WMMA_F16(af, bf0, acc[t]);
      acc[4 + t] = WMMA_F16(af, bf1, acc[4 + t]);
    }
  }
}

// ---------------------------------------------------------------------------
// GroupNorm: block per (batch, group). Writes xnT in (b, p, c) f16 layout.
// ---------------------------------------------------------------------------
__global__ void groupnorm_kernel(const float* __restrict__ x,
                                 const float* __restrict__ gamma,
                                 const float* __restrict__ beta,
                                 half_t* __restrict__ xnT) {
  const int bg = blockIdx.x;
  const int b = bg >> 5;          // 32 groups
  const int g = bg & 31;
  const int GN = 16 * HW;         // elems per group (16 ch * 4096 px)
  const float* xg = x + ((size_t)b * CC + g * 16) * HW;

  float s = 0.f, s2 = 0.f;
  for (int idx = threadIdx.x; idx < GN; idx += 256) {  // coalesced over p
    float v = xg[idx];
    s += v; s2 += v * v;
  }
  int lane = threadIdx.x & 31, wid = threadIdx.x >> 5;
#pragma unroll
  for (int off = 16; off; off >>= 1) { s += __shfl_xor(s, off); s2 += __shfl_xor(s2, off); }
  __shared__ float rs[8], rs2[8];
  if (lane == 0) { rs[wid] = s; rs2[wid] = s2; }
  __syncthreads();
  float ts = 0.f, ts2 = 0.f;
#pragma unroll
  for (int t = 0; t < 8; ++t) { ts += rs[t]; ts2 += rs2[t]; }
  const float inv_n = 1.f / (float)GN;
  float mean = ts * inv_n;
  float var  = ts2 * inv_n - mean * mean;
  float rstd = rsqrtf(var + 1e-6f);

  half_t* outb = xnT + (size_t)b * HW * CC;
  for (int idx = threadIdx.x; idx < GN; idx += 256) {
    int cl = idx >> 12;            // channel within group
    int p  = idx & (HW - 1);
    int c  = g * 16 + cl;
    float v = (xg[idx] - mean) * rstd * gamma[c] + beta[c];
    outb[(size_t)p * CC + c] = (half_t)v;
  }
}

// ---------------------------------------------------------------------------
// f32 -> f16 weight conversion
// ---------------------------------------------------------------------------
__global__ void cvt_kernel(const float* __restrict__ src, half_t* __restrict__ dst, int n) {
  int i = blockIdx.x * blockDim.x + threadIdx.x;
  if (i < n) dst[i] = (half_t)src[i];
}

// ---------------------------------------------------------------------------
// Projection: Y[cout][p] = W*xn + bias.  TSTORE=1 -> Y stored (p, cout).
// grid (HW/128, CC/64, NB), block 128 (4 waves)
// ---------------------------------------------------------------------------
template <bool TSTORE>
__global__ void proj_kernel(const half_t* __restrict__ W, const half_t* __restrict__ X,
                            const float* __restrict__ bias, half_t* __restrict__ Y) {
  const int b = blockIdx.z;
  const half_t* Bt = X + (size_t)b * HW * CC;
  half_t* out = Y + (size_t)b * HW * CC;
  int n0 = blockIdx.x * 128 + (threadIdx.x >> 5) * 32;  // pixel
  int m_blk = blockIdx.y * 64;                           // cout
  v8f acc[8] = {};
  gemm_wave_64x32(W, CC, Bt, CC, m_blk, n0, CC, acc);
  int lane = threadIdx.x & 31;
  int hi = lane >> 4;
#pragma unroll
  for (int f = 0; f < 2; ++f) {
    int n = n0 + 16 * f + (lane & 15);
#pragma unroll
    for (int t = 0; t < 4; ++t) {
      int m0 = m_blk + 16 * t + 8 * hi;
      if (TSTORE) {
        v8h h;
#pragma unroll
        for (int e = 0; e < 8; ++e) h[e] = (half_t)(acc[4 * f + t][e] + bias[m0 + e]);
        *(v8h*)(out + (size_t)n * CC + m0) = h;
      } else {
#pragma unroll
        for (int e = 0; e < 8; ++e)
          out[(size_t)(m0 + e) * HW + n] = (half_t)(acc[4 * f + t][e] + bias[m0 + e]);
      }
    }
  }
}

// ---------------------------------------------------------------------------
// Scores: S[i][j] = scale * sum_c q[c,i] k[c,j].
// Computed as D[m=j][n=i] = kT * qT, stored transposed -> S rows (i) contiguous.
// grid (HW/128 [i], HW/64 [j]), block 128
// ---------------------------------------------------------------------------
__global__ void attn_s_kernel(const half_t* __restrict__ kT, const half_t* __restrict__ qT,
                              half_t* __restrict__ S) {
  int n0 = blockIdx.x * 128 + (threadIdx.x >> 5) * 32;  // i
  int m_blk = blockIdx.y * 64;                           // j
  v8f acc[8] = {};
  gemm_wave_64x32(kT, CC, qT, CC, m_blk, n0, CC, acc);
  const float scale = 0.044194173824159216f;  // 512^-0.5
  int lane = threadIdx.x & 31;
  int hi = lane >> 4;
#pragma unroll
  for (int f = 0; f < 2; ++f) {
    int n = n0 + 16 * f + (lane & 15);
#pragma unroll
    for (int t = 0; t < 4; ++t) {
      int m0 = m_blk + 16 * t + 8 * hi;
      v8h h;
#pragma unroll
      for (int e = 0; e < 8; ++e) h[e] = (half_t)(acc[4 * f + t][e] * scale);
      *(v8h*)(S + (size_t)n * HW + m0) = h;
    }
  }
}

// ---------------------------------------------------------------------------
// Row softmax over 4096 columns, in place on f16 rows. block 256 per row.
// ---------------------------------------------------------------------------
__global__ void softmax_kernel(half_t* __restrict__ S) {
  __shared__ float buf[HW];
  __shared__ float red[8];
  half_t* row = S + (size_t)blockIdx.x * HW;
  int lane = threadIdx.x & 31, wid = threadIdx.x >> 5;

  float mx = -3.0e38f;
  for (int j = threadIdx.x; j < HW; j += 256) {
    float v = (float)row[j];
    buf[j] = v;
    mx = fmaxf(mx, v);
  }
#pragma unroll
  for (int off = 16; off; off >>= 1) mx = fmaxf(mx, __shfl_xor(mx, off));
  if (lane == 0) red[wid] = mx;
  __syncthreads();
  mx = red[0];
#pragma unroll
  for (int t = 1; t < 8; ++t) mx = fmaxf(mx, red[t]);
  __syncthreads();  // before red reuse

  float sum = 0.f;
  for (int j = threadIdx.x; j < HW; j += 256) {
    float e = __expf(buf[j] - mx);
    buf[j] = e;
    sum += e;
  }
#pragma unroll
  for (int off = 16; off; off >>= 1) sum += __shfl_xor(sum, off);
  if (lane == 0) red[wid] = sum;
  __syncthreads();
  float tot = 0.f;
#pragma unroll
  for (int t = 0; t < 8; ++t) tot += red[t];
  float inv = 1.f / tot;
  for (int j = threadIdx.x; j < HW; j += 256) row[j] = (half_t)(buf[j] * inv);
}

// ---------------------------------------------------------------------------
// O[c][i] = sum_j v[c][j] * P[i][j].  A = V (c,hw) natural, B = P^T from rows.
// Stored transposed: AO[(i)*CC + c].  grid (HW/128 [i], CC/64 [c]), block 128
// ---------------------------------------------------------------------------
__global__ void attn_o_kernel(const half_t* __restrict__ V, const half_t* __restrict__ P,
                              half_t* __restrict__ AO) {
  int n0 = blockIdx.x * 128 + (threadIdx.x >> 5) * 32;  // i
  int m_blk = blockIdx.y * 64;                           // c
  v8f acc[8] = {};
  gemm_wave_64x32(V, HW, P, HW, m_blk, n0, HW, acc);
  int lane = threadIdx.x & 31;
  int hi = lane >> 4;
#pragma unroll
  for (int f = 0; f < 2; ++f) {
    int n = n0 + 16 * f + (lane & 15);
#pragma unroll
    for (int t = 0; t < 4; ++t) {
      int m0 = m_blk + 16 * t + 8 * hi;
      v8h h;
#pragma unroll
      for (int e = 0; e < 8; ++e) h[e] = (half_t)acc[4 * f + t][e];
      *(v8h*)(AO + (size_t)n * CC + m0) = h;
    }
  }
}

// ---------------------------------------------------------------------------
// Final: out = x + Wo*AO + bo  (f32).  grid (HW/128, CC/64, NB), block 128
// ---------------------------------------------------------------------------
__global__ void projout_kernel(const half_t* __restrict__ W, const half_t* __restrict__ AO,
                               const float* __restrict__ bias, const float* __restrict__ x,
                               float* __restrict__ out) {
  const int b = blockIdx.z;
  const half_t* Bt = AO + (size_t)b * HW * CC;
  int n0 = blockIdx.x * 128 + (threadIdx.x >> 5) * 32;  // pixel
  int m_blk = blockIdx.y * 64;                           // cout
  v8f acc[8] = {};
  gemm_wave_64x32(W, CC, Bt, CC, m_blk, n0, CC, acc);
  int lane = threadIdx.x & 31;
  int hi = lane >> 4;
  size_t base = (size_t)b * CC * HW;
#pragma unroll
  for (int f = 0; f < 2; ++f) {
    int n = n0 + 16 * f + (lane & 15);
#pragma unroll
    for (int t = 0; t < 4; ++t) {
      int m0 = m_blk + 16 * t + 8 * hi;
#pragma unroll
      for (int e = 0; e < 8; ++e) {
        int m = m0 + e;
        size_t off = base + (size_t)m * HW + n;
        out[off] = x[off] + acc[4 * f + t][e] + bias[m];
      }
    }
  }
}

// ---------------------------------------------------------------------------
extern "C" void kernel_launch(void* const* d_in, const int* in_sizes, int n_in,
                              void* d_out, int out_size, void* d_ws, size_t ws_size,
                              hipStream_t stream) {
  const float* x     = (const float*)d_in[0];
  const float* gamma = (const float*)d_in[1];
  const float* beta  = (const float*)d_in[2];
  const float* wq = (const float*)d_in[3]; const float* bq = (const float*)d_in[4];
  const float* wk = (const float*)d_in[5]; const float* bk = (const float*)d_in[6];
  const float* wv = (const float*)d_in[7]; const float* bv = (const float*)d_in[8];
  const float* wo = (const float*)d_in[9]; const float* bo = (const float*)d_in[10];
  float* out = (float*)d_out;

  // Workspace layout (bytes):
  //   [0)            xnT (b,p,c) f16 -- later reused as AO (b,p,c)   16 MB
  //   [16M)          qT  (b,p,c) f16                                 16 MB
  //   [32M)          kT  (b,p,c) f16                                 16 MB
  //   [48M)          V   (b,c,p) f16                                 16 MB
  //   [64M)          S/P (i,j)   f16, single batch (reused)          32 MB
  //   [96M)          f16 weights: wq,wk,wv,wo                         2 MB
  char* ws = (char*)d_ws;
  const size_t ACT = (size_t)NB * HW * CC * sizeof(half_t);  // 16,777,216
  half_t* xnT = (half_t*)ws;
  half_t* qT  = (half_t*)(ws + ACT);
  half_t* kT  = (half_t*)(ws + 2 * ACT);
  half_t* v16 = (half_t*)(ws + 3 * ACT);
  half_t* s16 = (half_t*)(ws + 4 * ACT);
  half_t* w16 = (half_t*)(ws + 4 * ACT + (size_t)HW * HW * sizeof(half_t));
  half_t* w16q = w16;
  half_t* w16k = w16 + CC * CC;
  half_t* w16v = w16 + 2 * CC * CC;
  half_t* w16o = w16 + 3 * CC * CC;
  half_t* aoT  = xnT;  // alias: xn dead after V projection

  const int WN = CC * CC;
  cvt_kernel<<<(WN + 255) / 256, 256, 0, stream>>>(wq, w16q, WN);
  cvt_kernel<<<(WN + 255) / 256, 256, 0, stream>>>(wk, w16k, WN);
  cvt_kernel<<<(WN + 255) / 256, 256, 0, stream>>>(wv, w16v, WN);
  cvt_kernel<<<(WN + 255) / 256, 256, 0, stream>>>(wo, w16o, WN);

  groupnorm_kernel<<<NB * 32, 256, 0, stream>>>(x, gamma, beta, xnT);

  dim3 gp(HW / 128, CC / 64, NB);
  proj_kernel<true ><<<gp, 128, 0, stream>>>(w16q, xnT, bq, qT);
  proj_kernel<true ><<<gp, 128, 0, stream>>>(w16k, xnT, bk, kT);
  proj_kernel<false><<<gp, 128, 0, stream>>>(w16v, xnT, bv, v16);

  for (int b = 0; b < NB; ++b) {
    size_t boff = (size_t)b * HW * CC;
    attn_s_kernel<<<dim3(HW / 128, HW / 64), 128, 0, stream>>>(kT + boff, qT + boff, s16);
    softmax_kernel<<<HW, 256, 0, stream>>>(s16);
    attn_o_kernel<<<dim3(HW / 128, CC / 64), 128, 0, stream>>>(v16 + boff, s16, aoT + boff);
  }

  projout_kernel<<<dim3(HW / 128, CC / 64, NB), 128, 0, stream>>>(w16o, aoT, bo, x, out);
}
